// MultiHeadLinearAttention_3951369912945
// MI455X (gfx1250) — matrix operand order repaired and checked by reading, not compiled, not run on hardware
//
#include <hip/hip_runtime.h>
#include <hip/hip_bf16.h>

// ---------------------------------------------------------------------------
// MultiHeadLinearAttention for MI455X (gfx1250), bf16 WMMA pipeline.
//   B=4, S=4096, H=16, D=64, E=1024
// ---------------------------------------------------------------------------

#define B_ 4
#define H_ 16
#define D_ 64
#define S_ 4096
#define E_ 1024
#define EPS_ 1e-6f

typedef __bf16 bf16_t;
typedef __attribute__((ext_vector_type(16))) __bf16 v16bf;
typedef __attribute__((ext_vector_type(8)))  __bf16 v8bf;
typedef __attribute__((ext_vector_type(8)))  float  v8f;

__device__ __forceinline__ v8f v8f_zero() {
    v8f z;
#pragma unroll
    for (int i = 0; i < 8; ++i) z[i] = 0.0f;
    return z;
}

// D = A*B + C, bf16 inputs, f32 accumulate.  Emits v_wmma_f32_16x16x32_bf16.
__device__ __forceinline__ v8f wmma_bf16(v16bf a, v16bf b, v8f c) {
    return __builtin_amdgcn_wmma_f32_16x16x32_bf16(
        /*neg_a=*/false, a, /*neg_b=*/false, b,
        /*c_mod=*/(short)0, c, /*reuse_a=*/false, /*reuse_b=*/false);
}

// A fragment (16x32, M x K) from a bf16 row-major row pointer (already offset
// to the K-chunk start).  ISA layout: lanes 0-15 hold K 0..7 / 16..23,
// lanes 16-31 hold K 8..15 / 24..31.   lg = lane>>4.
__device__ __forceinline__ v16bf load_a_frag(const bf16_t* __restrict__ row, int lg) {
    const v8bf lo = *(const v8bf*)(row + lg * 8);
    const v8bf hi = *(const v8bf*)(row + lg * 8 + 16);
    v16bf a;
#pragma unroll
    for (int i = 0; i < 8; ++i) { a[i] = lo[i]; a[i + 8] = hi[i]; }
    return a;
}

// Same A fragment but sourced from f32 memory (converted to bf16 on the fly).
__device__ __forceinline__ v16bf load_a_frag_f32(const float* __restrict__ row, int lg) {
    const v8f lo = *(const v8f*)(row + lg * 8);
    const v8f hi = *(const v8f*)(row + lg * 8 + 16);
    v16bf a;
#pragma unroll
    for (int i = 0; i < 8; ++i) { a[i] = (bf16_t)lo[i]; a[i + 8] = (bf16_t)hi[i]; }
    return a;
}

// B fragment (32x16, K x N) from a transposed matrix: row pointer = column n
// of the logical B (contiguous in K).  lanes 0-15: K 0..7 / 16..23, lanes 16-31: K 8..15 / 24..31.
__device__ __forceinline__ v16bf load_b_frag(const bf16_t* __restrict__ colrow, int lg) {
    const v8bf lo = *(const v8bf*)(colrow + lg * 8);
    const v8bf hi = *(const v8bf*)(colrow + lg * 8 + 16);
    v16bf b;
#pragma unroll
    for (int i = 0; i < 8; ++i) { b[i] = lo[i]; b[i + 8] = hi[i]; }
    return b;
}

// ---------------------------------------------------------------------------
// Kernel 1: weight conversion.  WqT/WkT/WvT[h][e][d] = W*[h][d][e] (bf16),
//           WoT[eo][e] = Wo[e][eo] (bf16).
// ---------------------------------------------------------------------------
__global__ void mhla_convert(const float* __restrict__ Wq, const float* __restrict__ Wk,
                             const float* __restrict__ Wv, const float* __restrict__ Wo,
                             bf16_t* __restrict__ WqT, bf16_t* __restrict__ WkT,
                             bf16_t* __restrict__ WvT, bf16_t* __restrict__ WoT) {
    const int idx = blockIdx.x * 256 + threadIdx.x;
    if (idx < E_ * E_) {
        const int eo = idx / E_, e = idx % E_;
        WoT[idx] = (bf16_t)Wo[(size_t)e * E_ + eo];
    }
    if (idx < H_ * D_ * D_) {
        const int h = idx / (D_ * D_);
        const int e = (idx / D_) % D_;
        const int d = idx % D_;
        const size_t src = (size_t)h * D_ * D_ + (size_t)d * D_ + e;
        WqT[idx] = (bf16_t)Wq[src];
        WkT[idx] = (bf16_t)Wk[src];
        WvT[idx] = (bf16_t)Wv[src];
    }
}

// ---------------------------------------------------------------------------
// Kernel 2: per-head projections + phi.  One wave per 16-row s-tile.
//   phiQ [bh][s][d]   (row-major, A operand of ctx GEMM)
//   phiKT[bh][d][s]   (transposed, A operand of kv GEMM)
//   vT   [bh][e][s]   (transposed, B operand of kv GEMM)
// ---------------------------------------------------------------------------
__global__ void mhla_proj(const float* __restrict__ query,
                          const bf16_t* __restrict__ WqT, const bf16_t* __restrict__ WkT,
                          const bf16_t* __restrict__ WvT,
                          bf16_t* __restrict__ phiQ, bf16_t* __restrict__ phiKT,
                          bf16_t* __restrict__ vT) {
    const int wave = threadIdx.x >> 5;
    const int lane = threadIdx.x & 31;
    const int lr = lane & 15, lg = lane >> 4;
    const int gwave = blockIdx.x * (blockDim.x >> 5) + wave;
    const int stile = gwave & (S_ / 16 - 1);
    const int bh    = gwave >> 8;           // S_/16 == 256
    const int b = bh / H_, h = bh % H_;
    const int s0 = stile * 16;

    // x row for this lane: x[b,h,s,d] = query[b, s, h*D + d]
    const float* xrow = query + ((size_t)b * S_ + (s0 + lr)) * E_ + h * D_;
    const v16bf a0 = load_a_frag_f32(xrow,      lg);   // d 0..31
    const v16bf a1 = load_a_frag_f32(xrow + 32, lg);   // d 32..63

    const bf16_t* Wm[3] = { WqT + (size_t)h * D_ * D_,
                            WkT + (size_t)h * D_ * D_,
                            WvT + (size_t)h * D_ * D_ };
#pragma unroll
    for (int m = 0; m < 3; ++m) {
        const bf16_t* W = Wm[m];
#pragma unroll
        for (int nt = 0; nt < 4; ++nt) {
            const bf16_t* wcol = W + (size_t)(nt * 16 + lr) * D_;  // WT row = col e
            v8f c = v8f_zero();
            c = wmma_bf16(a0, load_b_frag(wcol,      lg), c);
            c = wmma_bf16(a1, load_b_frag(wcol + 32, lg), c);
            if (m < 2) {   // phi(x) = elu(x)+1 = x+1 (x>0) else exp(x)
#pragma unroll
                for (int r = 0; r < 8; ++r) {
                    const float x = c[r];
                    c[r] = (x > 0.0f) ? (x + 1.0f) : __expf(x);
                }
            }
            if (m == 0) {
                // phiQ[bh][s][d] : per-lane column fixed, scattered rows
#pragma unroll
                for (int r = 0; r < 8; ++r) {
                    const int s = s0 + r + 8 * lg;
                    phiQ[((size_t)bh * S_ + s) * D_ + nt * 16 + lr] = (bf16_t)c[r];
                }
            } else {
                // transposed store: lane's column is fixed, rows s are the
                // 8 consecutive accumulator VGPRs -> one packed 16B store
                bf16_t* dst = (m == 1) ? phiKT : vT;
                v8bf pk;
#pragma unroll
                for (int r = 0; r < 8; ++r) pk[r] = (bf16_t)c[r];
                *(v8bf*)(dst + ((size_t)bh * D_ + nt * 16 + lr) * S_ + s0 + 8 * lg) = pk;
            }
        }
    }
}

// ---------------------------------------------------------------------------
// Kernel 3: kv[b,h] = phiK^T . V  (64x64, K=S=4096) + k_sum.
// One 4-wave block per (b,h); wave w owns row-tile dm=w across all 4 col
// tiles (A-frag reused 4x), register double-buffered over K.
// Stores kvT[bh][e][d] bf16 (B operand of ctx GEMM).
// ---------------------------------------------------------------------------
__global__ void mhla_kv(const bf16_t* __restrict__ phiKT, const bf16_t* __restrict__ vT,
                        bf16_t* __restrict__ kvT, float* __restrict__ ksum) {
    const int bh = blockIdx.x;
    const int wave = threadIdx.x >> 5;        // == dm (0..3)
    const int lane = threadIdx.x & 31;
    const int lr = lane & 15, lg = lane >> 4;
    const int dm = wave;

    const bf16_t* Arow = phiKT + ((size_t)bh * D_ + dm * 16 + lr) * S_;
    const bf16_t* Br0  = vT + ((size_t)bh * D_ + 0 * 16 + lr) * S_;
    const bf16_t* Br1  = vT + ((size_t)bh * D_ + 1 * 16 + lr) * S_;
    const bf16_t* Br2  = vT + ((size_t)bh * D_ + 2 * 16 + lr) * S_;
    const bf16_t* Br3  = vT + ((size_t)bh * D_ + 3 * 16 + lr) * S_;

    v8f c0 = v8f_zero(), c1 = v8f_zero(), c2 = v8f_zero(), c3 = v8f_zero();

    // software pipeline: loads for step k+1 issued before WMMAs of step k
    v16bf a  = load_a_frag(Arow, lg);
    v16bf b0 = load_b_frag(Br0, lg);
    v16bf b1 = load_b_frag(Br1, lg);
    v16bf b2 = load_b_frag(Br2, lg);
    v16bf b3 = load_b_frag(Br3, lg);
    for (int kk = 0; kk < S_; kk += 32) {
        const int kn = (kk + 32 < S_) ? (kk + 32) : 0;   // last iter: dummy reload
        const v16bf an  = load_a_frag(Arow + kn, lg);
        const v16bf b0n = load_b_frag(Br0 + kn, lg);
        const v16bf b1n = load_b_frag(Br1 + kn, lg);
        const v16bf b2n = load_b_frag(Br2 + kn, lg);
        const v16bf b3n = load_b_frag(Br3 + kn, lg);
        c0 = wmma_bf16(a, b0, c0);
        c1 = wmma_bf16(a, b1, c1);
        c2 = wmma_bf16(a, b2, c2);
        c3 = wmma_bf16(a, b3, c3);
        a = an; b0 = b0n; b1 = b1n; b2 = b2n; b3 = b3n;
    }

    // kvT[e][d]: lane column e fixed, rows d consecutive across VGPRs -> packed
    v8f accs[4] = { c0, c1, c2, c3 };
#pragma unroll
    for (int en = 0; en < 4; ++en) {
        v8bf pk;
#pragma unroll
        for (int r = 0; r < 8; ++r) pk[r] = (bf16_t)accs[en][r];
        *(v8bf*)(kvT + ((size_t)bh * D_ + en * 16 + lr) * D_ + dm * 16 + 8 * lg) = pk;
    }

    // deterministic k_sum[d] = sum_s phiK[s][d] (one thread per d)
    if (threadIdx.x < D_) {
        const bf16_t* row = phiKT + ((size_t)bh * D_ + threadIdx.x) * S_;
        float acc = 0.0f;
        for (int s = 0; s < S_; s += 8) {
            const v8bf v = *(const v8bf*)(row + s);
#pragma unroll
            for (int i = 0; i < 8; ++i) acc += (float)v[i];
        }
        ksum[bh * D_ + threadIdx.x] = acc;
    }
}

// ---------------------------------------------------------------------------
// Kernel 4: ctx = (phiQ . kv) / (phiQ . k_sum + eps), write combined[B*S][E].
// One wave per 16-row s-tile.
// ---------------------------------------------------------------------------
__global__ void mhla_ctx(const bf16_t* __restrict__ phiQ, const bf16_t* __restrict__ kvT,
                         const float* __restrict__ ksum, bf16_t* __restrict__ comb) {
    __shared__ float zinv[8 * 16];
    const int wave = threadIdx.x >> 5;
    const int lane = threadIdx.x & 31;
    const int lr = lane & 15, lg = lane >> 4;
    const int gwave = blockIdx.x * (blockDim.x >> 5) + wave;
    const int stile = gwave & (S_ / 16 - 1);
    const int bh    = gwave >> 8;
    const int b = bh / H_, h = bh % H_;
    const int s0 = stile * 16;

    // z per row (lanes 0..15 each handle one of the 16 rows)
    if (lane < 16) {
        const bf16_t* qrow = phiQ + ((size_t)bh * S_ + s0 + lane) * D_;
        const float* ks = ksum + bh * D_;
        float z = EPS_;
#pragma unroll
        for (int d = 0; d < D_; ++d) z += (float)qrow[d] * ks[d];
        zinv[wave * 16 + lane] = 1.0f / z;
    }
    __syncthreads();

    const bf16_t* arow = phiQ + ((size_t)bh * S_ + s0 + lr) * D_;
    const v16bf a0 = load_a_frag(arow,      lg);
    const v16bf a1 = load_a_frag(arow + 32, lg);

#pragma unroll
    for (int nt = 0; nt < 4; ++nt) {
        const bf16_t* bcol = kvT + ((size_t)bh * D_ + nt * 16 + lr) * D_;
        v8f c = v8f_zero();
        c = wmma_bf16(a0, load_b_frag(bcol,      lg), c);
        c = wmma_bf16(a1, load_b_frag(bcol + 32, lg), c);
#pragma unroll
        for (int r = 0; r < 8; ++r) {
            const int s = s0 + r + 8 * lg;
            const float val = c[r] * zinv[wave * 16 + r + 8 * lg];
            comb[((size_t)b * S_ + s) * E_ + h * D_ + nt * 16 + lr] = (bf16_t)val;
        }
    }
}

// ---------------------------------------------------------------------------
// Kernel 5: out = combined @ Wo.  M = B*S = 16384, N = K = 1024.
// 8 waves/block; block covers 256 rows x 64 cols; each wave owns a 32x64 tile
// (8 accumulators, B-frags amortized over 2 row tiles), register
// double-buffered over K.
// ---------------------------------------------------------------------------
__global__ void mhla_outproj(const bf16_t* __restrict__ comb, const bf16_t* __restrict__ WoT,
                             float* __restrict__ out) {
    const int wave = threadIdx.x >> 5;
    const int lane = threadIdx.x & 31;
    const int lr = lane & 15, lg = lane >> 4;
    const int bm = blockIdx.x & 63;             // 16384/256 = 64
    const int bn = blockIdx.x >> 6;             // 1024/64  = 16
    const int m0 = bm * 256 + wave * 32;        // two 16-row tiles: m0, m0+16
    const int n0 = bn * 64;

    const bf16_t* arow0 = comb + (size_t)(m0 + lr) * E_;
    const bf16_t* arow1 = comb + (size_t)(m0 + 16 + lr) * E_;
    const bf16_t* bw0 = WoT + (size_t)(n0 + 0 * 16 + lr) * E_;
    const bf16_t* bw1 = WoT + (size_t)(n0 + 1 * 16 + lr) * E_;
    const bf16_t* bw2 = WoT + (size_t)(n0 + 2 * 16 + lr) * E_;
    const bf16_t* bw3 = WoT + (size_t)(n0 + 3 * 16 + lr) * E_;

    v8f acc[2][4];
#pragma unroll
    for (int i = 0; i < 2; ++i)
#pragma unroll
        for (int j = 0; j < 4; ++j) acc[i][j] = v8f_zero();

    // software pipeline over K
    v16bf a0 = load_a_frag(arow0, lg);
    v16bf a1 = load_a_frag(arow1, lg);
    v16bf b0 = load_b_frag(bw0, lg);
    v16bf b1 = load_b_frag(bw1, lg);
    v16bf b2 = load_b_frag(bw2, lg);
    v16bf b3 = load_b_frag(bw3, lg);
    for (int kk = 0; kk < E_; kk += 32) {
        const int kn = (kk + 32 < E_) ? (kk + 32) : 0;   // last iter: dummy reload
        __builtin_prefetch(arow0 + kn + 128, 0, 0);      // global_prefetch_b8
        const v16bf a0n = load_a_frag(arow0 + kn, lg);
        const v16bf a1n = load_a_frag(arow1 + kn, lg);
        const v16bf b0n = load_b_frag(bw0 + kn, lg);
        const v16bf b1n = load_b_frag(bw1 + kn, lg);
        const v16bf b2n = load_b_frag(bw2 + kn, lg);
        const v16bf b3n = load_b_frag(bw3 + kn, lg);
        acc[0][0] = wmma_bf16(a0, b0, acc[0][0]);
        acc[0][1] = wmma_bf16(a0, b1, acc[0][1]);
        acc[0][2] = wmma_bf16(a0, b2, acc[0][2]);
        acc[0][3] = wmma_bf16(a0, b3, acc[0][3]);
        acc[1][0] = wmma_bf16(a1, b0, acc[1][0]);
        acc[1][1] = wmma_bf16(a1, b1, acc[1][1]);
        acc[1][2] = wmma_bf16(a1, b2, acc[1][2]);
        acc[1][3] = wmma_bf16(a1, b3, acc[1][3]);
        a0 = a0n; a1 = a1n; b0 = b0n; b1 = b1n; b2 = b2n; b3 = b3n;
    }

#pragma unroll
    for (int i = 0; i < 2; ++i) {
#pragma unroll
        for (int r = 0; r < 8; ++r) {
            const size_t row = (size_t)(m0 + i * 16 + r + 8 * lg) * E_;
            out[row + n0 + 0 * 16 + lr] = acc[i][0][r];
            out[row + n0 + 1 * 16 + lr] = acc[i][1][r];
            out[row + n0 + 2 * 16 + lr] = acc[i][2][r];
            out[row + n0 + 3 * 16 + lr] = acc[i][3][r];
        }
    }
}

// ---------------------------------------------------------------------------
extern "C" void kernel_launch(void* const* d_in, const int* in_sizes, int n_in,
                              void* d_out, int out_size, void* d_ws, size_t ws_size,
                              hipStream_t stream) {
    const float* query = (const float*)d_in[0];
    const float* Wq    = (const float*)d_in[1];
    const float* Wk    = (const float*)d_in[2];
    const float* Wv    = (const float*)d_in[3];
    const float* Wo    = (const float*)d_in[4];
    float* out = (float*)d_out;

    // workspace carve-up (256B aligned slabs)
    char* ws = (char*)d_ws;
    size_t off = 0;
    auto take = [&](size_t bytes) -> char* {
        char* p = ws + off;
        off += (bytes + 255) & ~(size_t)255;
        return p;
    };
    bf16_t* WqT   = (bf16_t*)take((size_t)H_ * D_ * D_ * 2);
    bf16_t* WkT   = (bf16_t*)take((size_t)H_ * D_ * D_ * 2);
    bf16_t* WvT   = (bf16_t*)take((size_t)H_ * D_ * D_ * 2);
    bf16_t* WoT   = (bf16_t*)take((size_t)E_ * E_ * 2);
    float*  ksum  = (float*) take((size_t)B_ * H_ * D_ * 4);
    bf16_t* kvT   = (bf16_t*)take((size_t)B_ * H_ * D_ * D_ * 2);
    bf16_t* phiQ  = (bf16_t*)take((size_t)B_ * H_ * S_ * D_ * 2);
    bf16_t* phiKT = (bf16_t*)take((size_t)B_ * H_ * S_ * D_ * 2);
    bf16_t* vT    = (bf16_t*)take((size_t)B_ * H_ * S_ * D_ * 2);
    bf16_t* comb  = (bf16_t*)take((size_t)B_ * S_ * E_ * 2);

    // 1. weight conversion / transposition
    mhla_convert<<<(E_ * E_ + 255) / 256, 256, 0, stream>>>(Wq, Wk, Wv, Wo,
                                                            WqT, WkT, WvT, WoT);
    // 2. projections + phi (B*H*(S/16) waves / 8 waves per block)
    mhla_proj<<<(B_ * H_ * (S_ / 16)) / 8, 256, 0, stream>>>(query, WqT, WkT, WvT,
                                                             phiQ, phiKT, vT);
    // 3. kv + k_sum (one 4-wave block per (b,h))
    mhla_kv<<<B_ * H_, 128, 0, stream>>>(phiKT, vT, kvT, ksum);
    // 4. ctx + normalization + head combine
    mhla_ctx<<<(B_ * H_ * (S_ / 16)) / 8, 256, 0, stream>>>(phiQ, kvT, ksum, comb);
    // 5. output projection (64 m-blocks x 16 n-blocks)
    mhla_outproj<<<64 * 16, 256, 0, stream>>>(comb, WoT, out);
}
